// GPT2AttentionSubLayer_77197742178330
// MI455X (gfx1250) — matrix operand order repaired and checked by reading, not compiled, not run on hardware
//
#include <hip/hip_runtime.h>
#include <hip/hip_bf16.h>
#include <stdint.h>

// Problem constants (match reference)
#define B_ 2
#define T_ 2048
#define C_ 2048
#define H_ 16
#define D_ 128
#define M_ (B_ * T_)   // 4096 rows for all projections

typedef unsigned short u16;
typedef __attribute__((ext_vector_type(16))) __bf16 v16bf;
typedef __attribute__((ext_vector_type(8)))  float  v8f;
typedef __attribute__((ext_vector_type(4)))  uint32_t v4u;
typedef __attribute__((ext_vector_type(8)))  int32_t  v8i;
typedef __attribute__((ext_vector_type(4)))  int32_t  v4i;

union AFrag {
    v16bf v;
    uint4 q[2];   // 32 bytes = 16 bf16
};

__device__ __forceinline__ u16 f2bf(float f) {
    uint32_t u = __float_as_uint(f);
    u += 0x7FFFu + ((u >> 16) & 1u);   // round-to-nearest-even
    return (u16)(u >> 16);
}

__device__ __forceinline__ uint32_t pack2(float a, float b) {
#if __has_builtin(__builtin_amdgcn_cvt_pk_bf16_f32)
    typedef __attribute__((ext_vector_type(2))) __bf16 v2bf;
    v2bf t = __builtin_amdgcn_cvt_pk_bf16_f32(a, b);
    return __builtin_bit_cast(uint32_t, t);
#else
    return (uint32_t)f2bf(a) | ((uint32_t)f2bf(b) << 16);
#endif
}

// Async global->LDS copy (CDNA5): 16B per lane, tracked by ASYNCcnt.
__device__ __forceinline__ void async_copy_b128(void* lds, const void* g) {
    asm volatile("global_load_async_to_lds_b128 %0, %1, off"
                 :: "v"((uint32_t)(uintptr_t)lds),
                    "v"((uint64_t)(uintptr_t)g)
                 : "memory");
}
__device__ __forceinline__ void wait_async0() {
    asm volatile("s_wait_asynccnt 0x0" ::: "memory");
}

// ---------------------------------------------------------------------------
// GEMM: Out[M,2048] = A[M,2048] * W[2048,2048], bf16 WMMA, fp32 accum.
// EPI: 0 = fp32 row-major out; 1 = RoPE + bf16 [B,H,T,D]; 2 = bf16 [B,H,T,D]
// ABF16: A operand is already bf16 (row-major) -> async copy to LDS.
// ---------------------------------------------------------------------------
template<bool ABF16, int EPI>
__global__ __launch_bounds__(256)
void gemm_kernel(const void* __restrict__ Aptr, const float* __restrict__ W,
                 void* __restrict__ Out)
{
    constexpr int BM = 64, BN = 128, BK = 32;
    constexpr int LDA = BK + 8;   // ushort stride, 80B (16B aligned)
    constexpr int LDB = BK + 8;   // B stored transposed [n][k]

    __shared__ u16 sA[2][BM * LDA];
    __shared__ u16 sB[2][BN * LDB];
    __shared__ float sC[(EPI == 0) ? 4 : BM * (BN + 4)];

    const int tid  = threadIdx.x;
    const int lane = tid & 31;
    const int wid  = tid >> 5;
    const int hl   = lane >> 4;    // half-wave select
    const int lr   = lane & 15;
    const int wm   = wid & 3;      // 4 waves along M (16 rows each)
    const int wn   = wid >> 2;     // 2 waves along N (64 cols each)
    const int m0   = blockIdx.y * BM;
    const int n0   = blockIdx.x * BN;

    // W tile handled as 4x4 blocks per thread: 4 float4 loads, 4 b64 stores.
    const int kb = (tid >> 5) * 4;   // k sub-row 0..28
    const int nb = (tid & 31) * 4;   // n sub-col 0..124

    v8f acc[4];
    for (int j = 0; j < 4; ++j)
        for (int r = 0; r < 8; ++r) acc[j][r] = 0.f;

    const float* Af = (const float*)Aptr;
    const u16*   Ab = (const u16*)Aptr;

    float4 ra[2]; float4 rb[4];

    auto load_global = [&](int kk, int buf) {
        if constexpr (ABF16) {
            int row = tid >> 2, k8 = (tid & 3) * 8;
            async_copy_b128(&sA[buf][row * LDA + k8],
                            Ab + (size_t)(m0 + row) * C_ + kk + k8);
        } else {
            for (int i = 0; i < 2; ++i) {
                int f = tid + i * 256;
                int row = f >> 3, k = (f & 7) * 4;
                ra[i] = *(const float4*)(Af + (size_t)(m0 + row) * C_ + kk + k);
            }
        }
        for (int j = 0; j < 4; ++j)
            rb[j] = *(const float4*)(W + (size_t)(kk + kb + j) * 2048 + n0 + nb);
    };

    auto store_lds = [&](int buf) {
        if constexpr (!ABF16) {
            for (int i = 0; i < 2; ++i) {
                int f = tid + i * 256;
                int row = f >> 3, k = (f & 7) * 4;
                uint2 p;
                p.x = pack2(ra[i].x, ra[i].y);
                p.y = pack2(ra[i].z, ra[i].w);
                *(uint2*)(&sA[buf][row * LDA + k]) = p;
            }
        }
        for (int j2 = 0; j2 < 4; ++j2) {   // transpose 4x4 block, b64 stores
            uint2 p;
            p.x = pack2(((const float*)&rb[0])[j2], ((const float*)&rb[1])[j2]);
            p.y = pack2(((const float*)&rb[2])[j2], ((const float*)&rb[3])[j2]);
            *(uint2*)(&sB[buf][(nb + j2) * LDB + kb]) = p;
        }
    };

    load_global(0, 0);
    store_lds(0);
    if constexpr (ABF16) wait_async0();
    __syncthreads();

    const int nk = C_ / BK;   // 64 K-steps
    for (int kt = 0; kt < nk; ++kt) {
        const int cur = kt & 1;
        if (kt + 1 < nk) load_global((kt + 1) * BK, (kt + 1) & 1);
        if (kt + 2 < nk)   // pull the tile after next toward L2
            __builtin_prefetch(W + (size_t)((kt + 2) * BK + (tid >> 4)) * 2048 +
                               n0 + (tid & 15) * 8, 0, 0);

        // Batch all fragment loads, then issue WMMAs back-to-back.
        AFrag af;
        {
            const u16* ap = &sA[cur][(wm * 16 + lr) * LDA + hl * 8];
            af.q[0] = *(const uint4*)(ap);
            af.q[1] = *(const uint4*)(ap + 16);
        }
        AFrag bfr[4];
        for (int j = 0; j < 4; ++j) {
            const u16* bp = &sB[cur][(wn * 64 + j * 16 + lr) * LDB + hl * 8];
            bfr[j].q[0] = *(const uint4*)(bp);
            bfr[j].q[1] = *(const uint4*)(bp + 16);
        }
        for (int j = 0; j < 4; ++j)
            acc[j] = __builtin_amdgcn_wmma_f32_16x16x32_bf16(
                false, af.v, false, bfr[j].v, (short)0, acc[j], false, false);

        if (kt + 1 < nk) store_lds((kt + 1) & 1);
        if constexpr (ABF16) wait_async0();
        __syncthreads();
    }

    if constexpr (EPI == 0) {
        float* out = (float*)Out;
        for (int j = 0; j < 4; ++j) {
            int col = n0 + wn * 64 + j * 16 + lr;
            for (int r = 0; r < 8; ++r) {
                int row = m0 + wm * 16 + r + 8 * hl;
                out[(size_t)row * 2048 + col] = acc[j][r];
            }
        }
    } else {
        // Stage fp32 tile in LDS so RoPE can pair columns d and d+64.
        for (int j = 0; j < 4; ++j) {
            int col = wn * 64 + j * 16 + lr;
            for (int r = 0; r < 8; ++r) {
                int row = wm * 16 + r + 8 * hl;
                sC[row * (BN + 4) + col] = acc[j][r];
            }
        }
        __syncthreads();
        u16* out = (u16*)Out;
        const int head = blockIdx.x;   // BN==128 -> one head per N tile
        for (int p = tid; p < BM * 64; p += 256) {
            int row = p >> 6;
            int d   = p & 63;
            float x1 = sC[row * (BN + 4) + d];
            float x2 = sC[row * (BN + 4) + d + 64];
            int m = m0 + row;
            int b = m >> 11;          // / T_
            int t = m & (T_ - 1);
            float o1 = x1, o2 = x2;
            if constexpr (EPI == 1) {
                // inv_freq = 10000^(-d/64) = exp(-d * ln(1e4)/64)
                float ang = (float)t * __expf((float)d * -0.14391157f);
                float sn, cs;
                __sincosf(ang, &sn, &cs);
                o1 = x1 * cs - x2 * sn;
                o2 = x1 * sn + x2 * cs;
            }
            u16* dst = out + (((size_t)b * H_ + head) * T_ + t) * D_;
            dst[d]      = f2bf(o1);
            dst[d + 64] = f2bf(o2);
        }
    }
}

// ---------------------------------------------------------------------------
// Flash attention: per (128-row Q tile, b*H+h). 8 waves x 16 rows.
// S = Q K^T (bf16 WMMA), online softmax, O += P V (bf16 WMMA).
// K tile: Tensor Data Mover DMA (with hardware LDS padding) when available,
// else async global->LDS. V transposed via packed b32 stores.
// ---------------------------------------------------------------------------
__global__ __launch_bounds__(256)
void attn_kernel(const u16* __restrict__ Q, const u16* __restrict__ Kk,
                 const u16* __restrict__ V, u16* __restrict__ Y)
{
    constexpr int BQ  = 128, BKV = 64;
    constexpr int SKS = D_ + 8;     // 136 ushorts (272B; 256B row + 16B pad)
    constexpr int SVS = BKV + 8;    // 72 ushorts (144B, 16B aligned)

    __shared__ u16 sK [BKV * SKS];       // K tile, [tk][d]
    __shared__ u16 sVt[D_ * SVS];        // V tile transposed, [d][tk]
    __shared__ u16 sP [8 * 16 * SVS];    // per-wave P staging, [16][tk]

    const int tid  = threadIdx.x;
    const int lane = tid & 31;
    const int wid  = tid >> 5;
    const int hl   = lane >> 4;
    const int lr   = lane & 15;
    const int qbase = blockIdx.x * BQ;
    const int bh    = blockIdx.y;
    const int b     = bh >> 4;
    const int h     = bh & 15;

    const size_t base = (size_t)bh * T_ * D_;

    // Load this wave's Q fragments (16 rows x full D=128) once.
    AFrag qf[4];
    {
        const u16* qrow = Q + base + (size_t)(qbase + wid * 16 + lr) * D_;
        for (int kd = 0; kd < 4; ++kd) {
            qf[kd].q[0] = *(const uint4*)(qrow + kd * 32 + hl * 8);
            qf[kd].q[1] = *(const uint4*)(qrow + kd * 32 + 16 + hl * 8);
        }
    }

    v8f o[8];
    for (int j = 0; j < 8; ++j)
        for (int r = 0; r < 8; ++r) o[j][r] = 0.f;
    float rmax[8], rsum[8];
    for (int r = 0; r < 8; ++r) { rmax[r] = -1e30f; rsum[r] = 0.f; }

    const float sc = 0.08838834764831845f;   // 1/sqrt(128)
    const int nkt = (qbase + BQ) / BKV;      // causal: only tiles <= q range

    for (int kt = 0; kt < nkt; ++kt) {
        const int kv0 = kt * BKV;

#if __has_builtin(__builtin_amdgcn_tensor_load_to_lds)
        // K tile via TDM: one descriptor-driven DMA, hardware pads each
        // 256B row with 16B so the LDS stride matches SKS=136 ushorts.
        if (wid == 0) {
            uint64_t ga = (uint64_t)(uintptr_t)(Kk + base + (size_t)kv0 * D_);
            v4u g0;
            g0[0] = 1u;                                   // count=1, user mode
            g0[1] = (uint32_t)(uintptr_t)&sK[0];          // lds_addr (bytes)
            g0[2] = (uint32_t)ga;                         // global_addr[31:0]
            g0[3] = (uint32_t)((ga >> 32) & 0x01FFFFFFu)  // global_addr[56:32]
                    | (2u << 30);                         // type=2 (image)
            v8i g1;
            g1[0] = (int)((1u << 16)      // data_size = 2 bytes
                        | (1u << 20)      // pad_enable
                        | (5u << 22)      // pad_interval: every 64 DWORDs
                        | (3u << 25));    // pad_amount: 4 DWORDs (16B)
            g1[1] = (int)(128u << 16);    // tensor_dim0 = 128 (D)
            g1[2] = (int)(2048u << 16);   // tensor_dim1 = 2048 (T)
            g1[3] = (int)(128u << 16);    // tile_dim0 = 128
            g1[4] = 64;                   // tile_dim1 = 64, tile_dim2 = 0
            g1[5] = 128;                  // tensor_dim0_stride = 128
            g1[6] = 0;
            g1[7] = 0;                    // tensor_dim1_stride unused (2D)
            v4i z4; z4[0] = 0; z4[1] = 0; z4[2] = 0; z4[3] = 0;
            v8i z8;
            for (int i = 0; i < 8; ++i) z8[i] = 0;
            __builtin_amdgcn_tensor_load_to_lds(g0, g1, z4, z4, z8, 0);
        }
#else
        // Fallback: async global->LDS copies (ASYNCcnt).
        for (int i = 0; i < 4; ++i) {
            int f = tid + i * 256;
            int row = f >> 4, d8 = (f & 15) * 8;
            async_copy_b128(&sK[row * SKS + d8],
                            Kk + base + (size_t)(kv0 + row) * D_ + d8);
        }
#endif

        // V tile: load row pairs, then packed-b32 transposed stores.
        uint4 va[2], vb[2];
        for (int i = 0; i < 2; ++i) {
            int u = tid + i * 256;
            int r2 = (u >> 4) * 2;       // even row
            int d8 = (u & 15) * 8;
            va[i] = *(const uint4*)(V + base + (size_t)(kv0 + r2) * D_ + d8);
            vb[i] = *(const uint4*)(V + base + (size_t)(kv0 + r2 + 1) * D_ + d8);
        }
        for (int i = 0; i < 2; ++i) {
            int u = tid + i * 256;
            int r2 = (u >> 4) * 2;
            int d8 = (u & 15) * 8;
            const u16* ea = (const u16*)&va[i];
            const u16* eb = (const u16*)&vb[i];
            for (int j = 0; j < 8; ++j) {
                uint32_t p = (uint32_t)ea[j] | ((uint32_t)eb[j] << 16);
                *(uint32_t*)(&sVt[(d8 + j) * SVS + r2]) = p;
            }
        }
#if __has_builtin(__builtin_amdgcn_tensor_load_to_lds)
        if (wid == 0) __builtin_amdgcn_s_wait_tensorcnt(0);
#else
        wait_async0();
#endif
        __syncthreads();

        // S = Q K^T : per k-step load 4 B frags, then 4 independent WMMAs.
        v8f s[4];
        for (int j = 0; j < 4; ++j)
            for (int r = 0; r < 8; ++r) s[j][r] = 0.f;
        for (int kd = 0; kd < 4; ++kd) {
            AFrag bfr[4];
            for (int j = 0; j < 4; ++j) {
                const u16* bp = &sK[(j * 16 + lr) * SKS + kd * 32 + hl * 8];
                bfr[j].q[0] = *(const uint4*)(bp);
                bfr[j].q[1] = *(const uint4*)(bp + 16);
            }
            for (int j = 0; j < 4; ++j)
                s[j] = __builtin_amdgcn_wmma_f32_16x16x32_bf16(
                    false, qf[kd].v, false, bfr[j].v, (short)0, s[j], false, false);
        }

        // Scale + causal mask.
        const int qr0 = qbase + wid * 16;
        for (int j = 0; j < 4; ++j) {
            int col = kv0 + j * 16 + lr;
            for (int r = 0; r < 8; ++r) {
                int row = qr0 + r + 8 * hl;
                float vv = s[j][r] * sc;
                s[j][r] = (col <= row) ? vv : -1e30f;
            }
        }

        // Online softmax (row stats across 16-lane halves).
        for (int r = 0; r < 8; ++r) {
            float mx = fmaxf(fmaxf(s[0][r], s[1][r]), fmaxf(s[2][r], s[3][r]));
            for (int off = 1; off < 16; off <<= 1)
                mx = fmaxf(mx, __shfl_xor(mx, off, 32));
            float mnew = fmaxf(rmax[r], mx);
            float corr = __expf(rmax[r] - mnew);
            rmax[r] = mnew;
            float ls = 0.f;
            for (int j = 0; j < 4; ++j) {
                float p = __expf(s[j][r] - mnew);
                s[j][r] = p;
                ls += p;
            }
            for (int off = 1; off < 16; off <<= 1)
                ls += __shfl_xor(ls, off, 32);
            rsum[r] = rsum[r] * corr + ls;
            for (int j = 0; j < 8; ++j) o[j][r] *= corr;
        }

        // C-layout -> A-layout for P via per-wave LDS (DS ops in-order per wave).
        u16* pw = &sP[wid * 16 * SVS];
        for (int j = 0; j < 4; ++j)
            for (int r = 0; r < 8; ++r)
                pw[(r + 8 * hl) * SVS + j * 16 + lr] = f2bf(s[j][r]);

        // O += P V : per k-step, groups of 4 batched B frags -> 4 WMMAs.
        for (int ks = 0; ks < 2; ++ks) {
            AFrag pf;
            const u16* pp = &pw[lr * SVS + ks * 32 + hl * 8];
            pf.q[0] = *(const uint4*)(pp);
            pf.q[1] = *(const uint4*)(pp + 16);
            for (int g = 0; g < 2; ++g) {
                AFrag vfr[4];
                for (int j = 0; j < 4; ++j) {
                    const u16* vp =
                        &sVt[((g * 4 + j) * 16 + lr) * SVS + ks * 32 + hl * 8];
                    vfr[j].q[0] = *(const uint4*)(vp);
                    vfr[j].q[1] = *(const uint4*)(vp + 16);
                }
                for (int j = 0; j < 4; ++j)
                    o[g * 4 + j] = __builtin_amdgcn_wmma_f32_16x16x32_bf16(
                        false, pf.v, false, vfr[j].v, (short)0,
                        o[g * 4 + j], false, false);
            }
        }
        __syncthreads();   // protect sK/sVt before next tile load
    }

    // Write y (bf16) as [b*T+t][h*128+d] for the output projection GEMM.
    for (int j = 0; j < 8; ++j) {
        int d = h * 128 + j * 16 + lr;
        for (int r = 0; r < 8; ++r) {
            int t = qbase + wid * 16 + r + 8 * hl;
            Y[((size_t)b * T_ + t) * 2048 + d] = f2bf(o[j][r] / rsum[r]);
        }
    }
}

// ---------------------------------------------------------------------------
extern "C" void kernel_launch(void* const* d_in, const int* in_sizes, int n_in,
                              void* d_out, int out_size, void* d_ws, size_t ws_size,
                              hipStream_t stream) {
    const float* xq = (const float*)d_in[0];
    const float* xk = (const float*)d_in[1];
    const float* xv = (const float*)d_in[2];
    const float* wq = (const float*)d_in[3];
    const float* wk = (const float*)d_in[4];
    const float* wv = (const float*)d_in[5];
    const float* wo = (const float*)d_in[6];

    // Workspace: q,k,v (bf16 [B,H,T,D]) + y (bf16 [M,2048]) = 4 x 16 MB.
    u16* q = (u16*)d_ws;
    u16* k = q + (size_t)M_ * 2048;
    u16* v = k + (size_t)M_ * 2048;
    u16* y = v + (size_t)M_ * 2048;

    dim3 blk(256);
    dim3 gproj(2048 / 128, M_ / 64);   // 16 x 64

    gemm_kernel<false, 1><<<gproj, blk, 0, stream>>>(xq, wq, q);   // Q + RoPE
    gemm_kernel<false, 1><<<gproj, blk, 0, stream>>>(xk, wk, k);   // K + RoPE
    gemm_kernel<false, 2><<<gproj, blk, 0, stream>>>(xv, wv, v);   // V

    dim3 gattn(T_ / 128, B_ * H_);     // 16 x 32
    attn_kernel<<<gattn, blk, 0, stream>>>(q, k, v, y);

    gemm_kernel<true, 0><<<gproj, blk, 0, stream>>>(y, wo, (float*)d_out);
}
